// TinyMoEBlock_9199819948300
// MI455X (gfx1250) — matrix operand order repaired and checked by reading, not compiled, not run on hardware
//
#include <hip/hip_runtime.h>
#include <cstdint>

#define T_TOKENS 8192
#define HID 1024
#define INTER 2816
#define NEXP 8

typedef __bf16 bf16;
typedef __attribute__((ext_vector_type(16))) bf16 v16bf;
typedef __attribute__((ext_vector_type(8)))  bf16 v8bf;
typedef __attribute__((ext_vector_type(8)))  float v8f;

#define BM 128
#define BN 64
#define BK 32
#define LDA 40   // padded LDS stride (bf16 elems): 80B, 16B-aligned rows, reduces bank conflicts
#define LDB 40

__device__ __forceinline__ v8f wmma_bf16(v16bf a, v16bf b, v8f c) {
  // v_wmma_f32_16x16x32_bf16: (neg_a, A, neg_b, B, c_mod, C, reuse_a, reuse_b)
  return __builtin_amdgcn_wmma_f32_16x16x32_bf16(false, a, false, b, (short)0, c, false, false);
}

__device__ __forceinline__ void cvt16(const float4& f0, const float4& f1,
                                      const float4& f2, const float4& f3, bf16* dst) {
  v8bf b0, b1;
  b0[0]=(bf16)f0.x; b0[1]=(bf16)f0.y; b0[2]=(bf16)f0.z; b0[3]=(bf16)f0.w;
  b0[4]=(bf16)f1.x; b0[5]=(bf16)f1.y; b0[6]=(bf16)f1.z; b0[7]=(bf16)f1.w;
  b1[0]=(bf16)f2.x; b1[1]=(bf16)f2.y; b1[2]=(bf16)f2.z; b1[3]=(bf16)f2.w;
  b1[4]=(bf16)f3.x; b1[5]=(bf16)f3.y; b1[6]=(bf16)f3.z; b1[7]=(bf16)f3.w;
  ((v8bf*)dst)[0] = b0;
  ((v8bf*)dst)[1] = b1;
}

__device__ __forceinline__ void cvt8(const float4& f0, const float4& f1, bf16* dst) {
  v8bf b0;
  b0[0]=(bf16)f0.x; b0[1]=(bf16)f0.y; b0[2]=(bf16)f0.z; b0[3]=(bf16)f0.w;
  b0[4]=(bf16)f1.x; b0[5]=(bf16)f1.y; b0[6]=(bf16)f1.z; b0[7]=(bf16)f1.w;
  ((v8bf*)dst)[0] = b0;
}

// fast SiLU: x * sigmoid(x) with v_rcp_f32 instead of IEEE divide
__device__ __forceinline__ float fast_silu(float g) {
  return g * __builtin_amdgcn_rcpf(1.f + __expf(-g));
}

// ---------------- kernel 0: zero per-expert counters ----------------
__global__ void k_zero(int* __restrict__ cnt) {
  if (threadIdx.x < NEXP) cnt[threadIdx.x] = 0;
}

// ---------------- kernel 1: routing (logits, top-2, softmax, counts) ----------------
__global__ __launch_bounds__(256) void k_route(const float* __restrict__ x,
                                               const float* __restrict__ wg,
                                               int* __restrict__ cnt,
                                               int* __restrict__ tkIdx,
                                               float* __restrict__ tkW) {
  __shared__ __align__(16) float wls[NEXP * HID];   // 32 KB: whole gate matrix
  for (int i = threadIdx.x; i < NEXP * HID; i += blockDim.x) wls[i] = wg[i];
  __syncthreads();

  const int t = blockIdx.x * blockDim.x + threadIdx.x;
  float acc[NEXP];
#pragma unroll
  for (int e = 0; e < NEXP; ++e) acc[e] = 0.f;
  const float4* xv = (const float4*)(x + (size_t)t * HID);
  for (int h = 0; h < HID / 4; ++h) {
    const float4 v = xv[h];
#pragma unroll
    for (int e = 0; e < NEXP; ++e) {
      const float4 w = ((const float4*)(wls + e * HID))[h];  // broadcast, no bank conflict
      acc[e] += v.x * w.x + v.y * w.y + v.z * w.z + v.w * w.w;
    }
  }
  // top-2 (first occurrence wins ties, matching lax.top_k)
  float v0 = -3.4e38f, v1 = -3.4e38f; int i0 = 0, i1 = 0;
#pragma unroll
  for (int e = 0; e < NEXP; ++e) {
    const float v = acc[e];
    if (v > v0) { v1 = v0; i1 = i0; v0 = v; i0 = e; }
    else if (v > v1) { v1 = v; i1 = e; }
  }
  const float w0 = 1.f / (1.f + __expf(v1 - v0));   // softmax over {v0, v1}
  const float w1 = 1.f - w0;
  tkIdx[2 * t] = i0; tkIdx[2 * t + 1] = i1;
  tkW[2 * t] = w0;   tkW[2 * t + 1] = w1;
  atomicAdd(&cnt[i0], 1);
  atomicAdd(&cnt[i1], 1);
}

// ---------------- kernel 2: exclusive scan over 8 counts ----------------
__global__ void k_scan(const int* __restrict__ cnt, int* __restrict__ off, int* __restrict__ cnt2) {
  if (threadIdx.x == 0) {
    int s = 0;
    for (int e = 0; e < NEXP; ++e) { off[e] = s; s += cnt[e]; cnt2[e] = 0; }
  }
}

// ---------------- kernel 3: compact token lists per expert ----------------
__global__ __launch_bounds__(256) void k_fill(const int* __restrict__ tkIdx,
                                              const float* __restrict__ tkW,
                                              const int* __restrict__ off,
                                              int* __restrict__ cnt2,
                                              int* __restrict__ tmap,
                                              float* __restrict__ wmapv,
                                              int* __restrict__ slotOf) {
  const int t = blockIdx.x * blockDim.x + threadIdx.x;
#pragma unroll
  for (int k = 0; k < 2; ++k) {
    const int e = tkIdx[2 * t + k];
    const float w = tkW[2 * t + k];
    const int pos = atomicAdd(&cnt2[e], 1);
    const int slot = off[e] + pos;
    tmap[slot] = t;
    wmapv[slot] = w;
    slotOf[2 * t + k] = slot;
  }
}

// ---------------- kernel 4: grouped gate/up GEMM + fused SwiGLU (double-buffered) ----------------
// C1[s,i] = x[tok(s)] . wgp[e][i,:],  C2 likewise with wup;  Aws[s,i] = silu(C1)*C2 (bf16)
__global__ __launch_bounds__(256) void k_gateup(const float* __restrict__ x,
                                                const float* __restrict__ wgp,
                                                const float* __restrict__ wup,
                                                const int* __restrict__ cnt,
                                                const int* __restrict__ off,
                                                const int* __restrict__ tmap,
                                                bf16* __restrict__ Aws) {
  const int e = blockIdx.z;
  const int ce = cnt[e];
  const int m0 = blockIdx.x * BM;
  if (m0 >= ce) return;
  const int n0 = blockIdx.y * BN;
  const int base = off[e];
  const int tid = threadIdx.x;

  __shared__ __align__(16) bf16 As[2][BM * LDA];
  __shared__ __align__(16) bf16 Bg[2][BN * LDB];
  __shared__ __align__(16) bf16 Bu[2][BN * LDB];
  __shared__ int Ts[BM];

  if (tid < BM) {
    const int r = m0 + tid;
    Ts[tid] = (r < ce) ? tmap[base + r] : tmap[base];   // safe gather for pad rows
  }
  __syncthreads();

  const int lane = tid & 31, wv = tid >> 5;
  const int wm = (wv & 3) * 32, wn = (wv >> 2) * 32;    // 4x2 waves -> 128x64 tile
  const int half = lane >> 4, ln = lane & 15;

  v8f accg[2][2], accu[2][2];
#pragma unroll
  for (int mi = 0; mi < 2; ++mi)
#pragma unroll
    for (int ni = 0; ni < 2; ++ni) { accg[mi][ni] = (v8f)(0.f); accu[mi][ni] = (v8f)(0.f); }

  const float* Wg0 = wgp + (size_t)e * INTER * HID + (size_t)n0 * HID;
  const float* Wu0 = wup + (size_t)e * INTER * HID + (size_t)n0 * HID;

  const int lr = tid >> 1;            // A loader: row 0..127
  const int lch = (tid & 1) * 16;     // col half
  const int br = (tid & 127) >> 1;    // B loader: row 0..63 (128 threads per matrix)
  const float* Wsel = (tid < 128) ? Wg0 : Wu0;
  bf16 (*Bsel)[BN * LDB] = (tid < 128) ? Bg : Bu;
  const float* Arow = x + (size_t)Ts[lr] * HID + lch;
  const float* Brow = Wsel + (size_t)br * HID + lch;

  float4 sa0, sa1, sa2, sa3;   // staged A (next K-step)
  float4 sb0, sb1, sb2, sb3;   // staged B (next K-step)

  // prologue: stage k0 = 0 into buffer 0
  { const float4* s = (const float4*)(Arow); sa0=s[0]; sa1=s[1]; sa2=s[2]; sa3=s[3]; }
  { const float4* s = (const float4*)(Brow); sb0=s[0]; sb1=s[1]; sb2=s[2]; sb3=s[3]; }
  cvt16(sa0, sa1, sa2, sa3, &As[0][lr * LDA + lch]);
  cvt16(sb0, sb1, sb2, sb3, &Bsel[0][br * LDB + lch]);
  __syncthreads();

  int cur = 0;
  for (int k0 = 0; k0 < HID; k0 += BK) {
    // prefetch next K-step while this one computes (hides VMEM latency behind WMMA)
    if (k0 + BK < HID) {
      const float4* s = (const float4*)(Arow + k0 + BK); sa0=s[0]; sa1=s[1]; sa2=s[2]; sa3=s[3];
      const float4* t_ = (const float4*)(Brow + k0 + BK); sb0=t_[0]; sb1=t_[1]; sb2=t_[2]; sb3=t_[3];
    }

    v16bf af[2];
#pragma unroll
    for (int mi = 0; mi < 2; ++mi) {
      // A 16x32 bf16 operand, this code's k order (matches B): 16 contiguous K at 16*half, M = lane&15
      const bf16* p = &As[cur][(wm + mi * 16 + ln) * LDA + 16 * half];
      v8bf* ap = (v8bf*)&af[mi];
      ap[0] = *(const v8bf*)p;
      ap[1] = *(const v8bf*)(p + 8);
    }
#pragma unroll
    for (int ni = 0; ni < 2; ++ni) {
      // B 32x16 bf16 operand, this code's k order (matches A): 16 contiguous K at 16*half, N = lane&15
      v16bf bgf, buf_;
      {
        const bf16* p = &Bg[cur][(wn + ni * 16 + ln) * LDB + 16 * half];
        v8bf* bp = (v8bf*)&bgf; bp[0] = *(const v8bf*)p; bp[1] = *(const v8bf*)(p + 8);
      }
      {
        const bf16* p = &Bu[cur][(wn + ni * 16 + ln) * LDB + 16 * half];
        v8bf* bp = (v8bf*)&buf_; bp[0] = *(const v8bf*)p; bp[1] = *(const v8bf*)(p + 8);
      }
#pragma unroll
      for (int mi = 0; mi < 2; ++mi) {
        accg[mi][ni] = wmma_bf16(af[mi], bgf, accg[mi][ni]);
        accu[mi][ni] = wmma_bf16(af[mi], buf_, accu[mi][ni]);
      }
    }
    __syncthreads();                       // all reads of buf[cur] done
    if (k0 + BK < HID) {
      cvt16(sa0, sa1, sa2, sa3, &As[cur ^ 1][lr * LDA + lch]);
      cvt16(sb0, sb1, sb2, sb3, &Bsel[cur ^ 1][br * LDB + lch]);
    }
    __syncthreads();                       // buf[nxt] ready
    cur ^= 1;
  }

  // fused SwiGLU, write bf16 intermediate.  C/D layout: M = r + 8*half, N = lane&15
#pragma unroll
  for (int mi = 0; mi < 2; ++mi)
#pragma unroll
    for (int ni = 0; ni < 2; ++ni)
#pragma unroll
      for (int r = 0; r < 8; ++r) {
        const int row = wm + mi * 16 + 8 * half + r;
        if (m0 + row < ce) {
          const float g = accg[mi][ni][r];
          const float u = accu[mi][ni][r];
          Aws[(size_t)(base + m0 + row) * INTER + (n0 + wn + ni * 16 + ln)] = (bf16)(fast_silu(g) * u);
        }
      }
}

// ---------------- kernel 5: grouped down GEMM, scaled by routing weight (double-buffered) ----------------
__global__ __launch_bounds__(256) void k_down(const bf16* __restrict__ Aws,
                                              const float* __restrict__ wdp,
                                              const int* __restrict__ cnt,
                                              const int* __restrict__ off,
                                              const float* __restrict__ wmapv,
                                              float* __restrict__ Dout) {
  const int e = blockIdx.z;
  const int ce = cnt[e];
  const int m0 = blockIdx.x * BM;
  if (m0 >= ce) return;
  const int h0 = blockIdx.y * BN;
  const int base = off[e];
  const int tid = threadIdx.x;

  __shared__ __align__(16) bf16 As[2][BM * LDA];
  __shared__ __align__(16) bf16 Bd[2][BN * LDB];
  __shared__ float Wsw[BM];

  if (tid < BM) {
    const int r = m0 + tid;
    Wsw[tid] = (r < ce) ? wmapv[base + r] : 0.f;
  }

  const int lane = tid & 31, wv = tid >> 5;
  const int wm = (wv & 3) * 32, wn = (wv >> 2) * 32;
  const int half = lane >> 4, ln = lane & 15;

  v8f acc[2][2];
#pragma unroll
  for (int mi = 0; mi < 2; ++mi)
#pragma unroll
    for (int ni = 0; ni < 2; ++ni) acc[mi][ni] = (v8f)(0.f);

  const float* Wd0 = wdp + (size_t)e * HID * INTER + (size_t)h0 * INTER;

  const int lr = tid >> 1;            // A loader: slot row 0..127
  const int lch = (tid & 1) * 16;
  int arow = m0 + lr; if (arow >= ce) arow = ce - 1;  // clamp: pad rows read valid memory
  const int dr = tid >> 2;            // B loader: h row 0..63
  const int dch = (tid & 3) * 8;
  const bf16* ArowP = Aws + (size_t)(base + arow) * INTER + lch;
  const float* BrowP = Wd0 + (size_t)dr * INTER + dch;

  v8bf saA0, saA1;                    // staged A (bf16 pass-through)
  float4 sbB0, sbB1;                  // staged B (f32 -> bf16 at store)

  { const v8bf* s = (const v8bf*)(ArowP); saA0 = s[0]; saA1 = s[1]; }
  { const float4* s = (const float4*)(BrowP); sbB0 = s[0]; sbB1 = s[1]; }
  ((v8bf*)&As[0][lr * LDA + lch])[0] = saA0;
  ((v8bf*)&As[0][lr * LDA + lch])[1] = saA1;
  cvt8(sbB0, sbB1, &Bd[0][dr * LDB + dch]);
  __syncthreads();

  int cur = 0;
  for (int k0 = 0; k0 < INTER; k0 += BK) {
    if (k0 + BK < INTER) {
      const v8bf* s = (const v8bf*)(ArowP + k0 + BK); saA0 = s[0]; saA1 = s[1];
      const float4* t_ = (const float4*)(BrowP + k0 + BK); sbB0 = t_[0]; sbB1 = t_[1];
    }

    v16bf af[2];
#pragma unroll
    for (int mi = 0; mi < 2; ++mi) {
      const bf16* p = &As[cur][(wm + mi * 16 + ln) * LDA + 16 * half];
      v8bf* ap = (v8bf*)&af[mi];
      ap[0] = *(const v8bf*)p;
      ap[1] = *(const v8bf*)(p + 8);
    }
#pragma unroll
    for (int ni = 0; ni < 2; ++ni) {
      v16bf bf_;
      const bf16* p = &Bd[cur][(wn + ni * 16 + ln) * LDB + 16 * half];
      v8bf* bp = (v8bf*)&bf_; bp[0] = *(const v8bf*)p; bp[1] = *(const v8bf*)(p + 8);
#pragma unroll
      for (int mi = 0; mi < 2; ++mi)
        acc[mi][ni] = wmma_bf16(af[mi], bf_, acc[mi][ni]);
    }
    __syncthreads();
    if (k0 + BK < INTER) {
      ((v8bf*)&As[cur ^ 1][lr * LDA + lch])[0] = saA0;
      ((v8bf*)&As[cur ^ 1][lr * LDA + lch])[1] = saA1;
      cvt8(sbB0, sbB1, &Bd[cur ^ 1][dr * LDB + dch]);
    }
    __syncthreads();
    cur ^= 1;
  }

#pragma unroll
  for (int mi = 0; mi < 2; ++mi)
#pragma unroll
    for (int ni = 0; ni < 2; ++ni)
#pragma unroll
      for (int r = 0; r < 8; ++r) {
        const int row = wm + mi * 16 + 8 * half + r;
        if (m0 + row < ce) {
          const float w = Wsw[row];
          Dout[(size_t)(base + m0 + row) * HID + (h0 + wn + ni * 16 + ln)] = w * acc[mi][ni][r];
        }
      }
}

// ---------------- kernel 6: deterministic combine (no float atomics) ----------------
__global__ __launch_bounds__(256) void k_combine(const float* __restrict__ Dout,
                                                 const int* __restrict__ slotOf,
                                                 float* __restrict__ out) {
  const int idx = blockIdx.x * blockDim.x + threadIdx.x;  // T*H/4 threads
  const int t = idx >> 8;            // H/4 == 256
  const int c = (idx & 255) << 2;
  const int s0 = slotOf[2 * t], s1 = slotOf[2 * t + 1];
  const float4 a = *(const float4*)(Dout + (size_t)s0 * HID + c);
  const float4 b = *(const float4*)(Dout + (size_t)s1 * HID + c);
  *(float4*)(out + (size_t)t * HID + c) = make_float4(a.x + b.x, a.y + b.y, a.z + b.z, a.w + b.w);
}

extern "C" void kernel_launch(void* const* d_in, const int* in_sizes, int n_in,
                              void* d_out, int out_size, void* d_ws, size_t ws_size,
                              hipStream_t stream) {
  const float* x   = (const float*)d_in[0];
  const float* wg  = (const float*)d_in[1];
  const float* wgp = (const float*)d_in[2];
  const float* wup = (const float*)d_in[3];
  const float* wdp = (const float*)d_in[4];
  float* out = (float*)d_out;

  char* ws = (char*)d_ws;
  size_t o = 0;
  auto alloc = [&](size_t bytes) { size_t r = o; o = (o + bytes + 255) & ~(size_t)255; return r; };

  const size_t SLOTS = (size_t)2 * T_TOKENS;
  int*   cnt    = (int*)  (ws + alloc(NEXP * sizeof(int)));
  int*   cnt2   = (int*)  (ws + alloc(NEXP * sizeof(int)));
  int*   off    = (int*)  (ws + alloc(NEXP * sizeof(int)));
  int*   tkIdx  = (int*)  (ws + alloc(SLOTS * sizeof(int)));
  float* tkW    = (float*)(ws + alloc(SLOTS * sizeof(float)));
  int*   tmap   = (int*)  (ws + alloc(SLOTS * sizeof(int)));
  float* wmapv  = (float*)(ws + alloc(SLOTS * sizeof(float)));
  int*   slotOf = (int*)  (ws + alloc(SLOTS * sizeof(int)));
  bf16*  Aws    = (bf16*) (ws + alloc(SLOTS * (size_t)INTER * sizeof(bf16)));   // ~92 MB
  float* Dout   = (float*)(ws + alloc(SLOTS * (size_t)HID * sizeof(float)));    // ~67 MB

  k_zero<<<1, 32, 0, stream>>>(cnt);
  k_route<<<T_TOKENS / 256, 256, 0, stream>>>(x, wg, cnt, tkIdx, tkW);
  k_scan<<<1, 1, 0, stream>>>(cnt, off, cnt2);
  k_fill<<<T_TOKENS / 256, 256, 0, stream>>>(tkIdx, tkW, off, cnt2, tmap, wmapv, slotOf);
  k_gateup<<<dim3(T_TOKENS / BM, INTER / BN, NEXP), 256, 0, stream>>>(x, wgp, wup, cnt, off, tmap, Aws);
  k_down<<<dim3(T_TOKENS / BM, HID / BN, NEXP), 256, 0, stream>>>(Aws, wdp, cnt, off, wmapv, Dout);
  k_combine<<<(T_TOKENS * (HID / 4)) / 256, 256, 0, stream>>>(Dout, slotOf, out);
}
